// SpeechRecognition_minGRUModel_61375082660004
// MI455X (gfx1250) — compile-verified, matrix operand order checked by reading, not run on hardware
//
#include <hip/hip_runtime.h>
#include <math.h>
#include <stdint.h>

// ---------------- model dims ----------------
#define BATCH   16
#define F0      128
#define T0      2048
#define C1      32
#define F1      64
#define T1      1024
#define M_TOK   (BATCH * T1)     // 16384 tokens
#define CF      (C1 * F1)        // 2048 flattened CNN features
#define DMODEL  512
#define NCLASS  29

typedef __attribute__((ext_vector_type(16))) __bf16 v16bf;
typedef __attribute__((ext_vector_type(8)))  __bf16 v8bf;
typedef __attribute__((ext_vector_type(8)))  float  v8f;
typedef int v4i __attribute__((vector_size(16)));   // matches async-LDS builtin param type

#ifndef __has_builtin
#define __has_builtin(x) 0
#endif

#if __has_builtin(__builtin_amdgcn_global_load_async_to_lds_b128)
#define GEMM_ASYNC_LDS 1
#else
#define GEMM_ASYNC_LDS 0
#endif

// ---------------- device math helpers ----------------
__device__ __forceinline__ float gelu_f(float x) {
  // jax.nn.gelu default (tanh approximation)
  float x3 = x * x * x;
  return 0.5f * x * (1.0f + tanhf(0.7978845608028654f * (x + 0.044715f * x3)));
}
__device__ __forceinline__ float softplus_f(float x) {
  return fmaxf(x, 0.0f) + log1pf(__expf(-fabsf(x)));
}
__device__ __forceinline__ float logaddexp_f(float a, float b) {
  return fmaxf(a, b) + log1pf(__expf(-fabsf(a - b)));
}
__device__ __forceinline__ float log_g_f(float h) {
  return (h >= 0.0f) ? __logf(h + 0.5f) : -softplus_f(-h);
}

// ---------------- WMMA GEMM:  C[M,N] = A[M,K] * W[N,K]^T (+bias)(+act)(+residual) ----------------
// A, W bf16 row-major (K contiguous).  Block tile 64x128 (8 waves, wave tile 32x32).
// A block-tile staged in LDS via gfx1250 async LDS loads, double buffered.

#define SMA_STRIDE 40   // bf16 elems per staged row: 32 K + 8 pad = 80 B (16B-aligned, bank-spread)

__device__ __forceinline__ void stage_a_chunk(const __bf16* gsrc, __bf16* ldst) {
#if GEMM_ASYNC_LDS
  __builtin_amdgcn_global_load_async_to_lds_b128(
      (__attribute__((address_space(1))) v4i*)(uintptr_t)gsrc,
      (__attribute__((address_space(3))) v4i*)(uintptr_t)ldst,
      /*imm offset=*/0, /*cpol=*/0);
#else
  *(uint4*)ldst = *(const uint4*)gsrc;   // fallback: load + ds_store
#endif
}
__device__ __forceinline__ void wait_async_le1() {
#if GEMM_ASYNC_LDS
#if __has_builtin(__builtin_amdgcn_s_wait_asynccnt)
  __builtin_amdgcn_s_wait_asynccnt(1);
#else
  asm volatile("s_wait_asynccnt 0x1" ::: "memory");
#endif
#endif
}
__device__ __forceinline__ void wait_async_le0() {
#if GEMM_ASYNC_LDS
#if __has_builtin(__builtin_amdgcn_s_wait_asynccnt)
  __builtin_amdgcn_s_wait_asynccnt(0);
#else
  asm volatile("s_wait_asynccnt 0x0" ::: "memory");
#endif
#endif
}

// fragment = 16 bf16 per lane: elems [0..7] at +0, [8..15] at +16 elements (per ISA 16-bit layout)
__device__ __forceinline__ v16bf load_frag_bf16(const __bf16* p) {
  v8bf lo = *(const v8bf*)(p);
  v8bf hi = *(const v8bf*)(p + 16);
  v16bf f;
#pragma unroll
  for (int i = 0; i < 8; ++i) { f[i] = lo[i]; f[i + 8] = hi[i]; }
  return f;
}

__device__ __forceinline__ v8f wmma_bf16(v16bf a, v16bf b, v8f c) {
  return __builtin_amdgcn_wmma_f32_16x16x32_bf16(
      /*neg_a=*/false, a, /*neg_b=*/false, b,
      /*c_mod=*/(short)0, c, /*reuse_a=*/false, /*reuse_b=*/false);
}

__device__ __forceinline__ void gemm_epilogue_tile(
    const v8f& c, int mBase, int nBase, const float* bias,
    float* outF, __bf16* outBf, int N, int ldc, int flags, int lane) {
  int n = nBase + (lane & 15);
  if (n >= N) return;
  int mOff = (lane & 16) ? 8 : 0;
  float bv = bias ? bias[n] : 0.0f;
#pragma unroll
  for (int j = 0; j < 8; ++j) {
    float acc = c[j] + bv;
    if (flags & 1) acc = gelu_f(acc);
    size_t idx = (size_t)(mBase + j + mOff) * (size_t)ldc + (size_t)n;
    if (flags & 2) acc += outF[idx];         // residual accumulate (read-modify-write, 1:1)
    if (outF)  outF[idx]  = acc;
    if (outBf) outBf[idx] = (__bf16)acc;
  }
}

__global__ __launch_bounds__(256) void gemm_bf16_kernel(
    const __bf16* __restrict__ A, const __bf16* __restrict__ W,
    const float* __restrict__ bias, float* __restrict__ outF,
    __bf16* __restrict__ outBf, int M, int N, int K, int ldc, int flags) {
  __shared__ __bf16 smA[2][64 * SMA_STRIDE];

  const int tid  = threadIdx.x;
  const int wave = tid >> 5;
  const int lane = tid & 31;
  const int wm = wave & 1;          // 2 waves along M
  const int wn = wave >> 1;         // 4 waves along N
  const int mBlock = blockIdx.y * 64;
  const int m0 = mBlock + wm * 32;
  const int n0 = blockIdx.x * 128 + wn * 32;

  const int laneLo = lane & 15;
  const int kSel   = (lane >> 4) & 1;   // hi lane group handles k+8 / k+24 pairs

  // ---- A staging assignment: 256 threads x 16B = 64 rows x 32 K bf16 ----
  const int sRow = tid >> 2;            // 0..63
  const int sCol = (tid & 3) * 8;       // 16B chunk (8 bf16)
  const __bf16* gA = A + (size_t)(mBlock + sRow) * K + sCol;
  __bf16* lA0 = &smA[0][sRow * SMA_STRIDE + sCol];
  __bf16* lA1 = &smA[1][sRow * SMA_STRIDE + sCol];

  // ---- B (weight) rows, direct from global (small, cache-hot) ----
  int rb0 = n0 + laneLo;      if (rb0 >= N) rb0 = N - 1;   // clamp OOB weight rows
  int rb1 = n0 + 16 + laneLo; if (rb1 >= N) rb1 = N - 1;
  const __bf16* pb0 = W + (size_t)rb0 * K;
  const __bf16* pb1 = W + (size_t)rb1 * K;

  // ---- A fragment LDS read offsets ----
  const int fr0 = (wm * 32 + laneLo) * SMA_STRIDE + kSel * 8;
  const int fr1 = fr0 + 16 * SMA_STRIDE;

  v8f c00 = {}, c01 = {}, c10 = {}, c11 = {};

  stage_a_chunk(gA, lA0);               // prologue fill of buffer 0
  int buf = 0;
  for (int k = 0; k < K; k += 32) {
    if (k + 32 < K) {
      stage_a_chunk(gA + (k + 32), buf ? lA0 : lA1);  // async fill of next buffer
      wait_async_le1();                 // own fill of current buffer retired
    } else {
      wait_async_le0();
    }
    __syncthreads();                    // all waves' fills of current buffer visible

    const __bf16* sb = smA[buf];
    v16bf a0 = load_frag_bf16(sb + fr0);
    v16bf a1 = load_frag_bf16(sb + fr1);
    const int kb = k + kSel * 8;
    if (k + 64 < K) {                   // gfx1250 global_prefetch_b8 path for weights
      __builtin_prefetch(pb0 + kb + 64, 0, 1);
      __builtin_prefetch(pb1 + kb + 64, 0, 1);
    }
    v16bf b0 = load_frag_bf16(pb0 + kb);
    v16bf b1 = load_frag_bf16(pb1 + kb);
    c00 = wmma_bf16(a0, b0, c00);
    c01 = wmma_bf16(a0, b1, c01);
    c10 = wmma_bf16(a1, b0, c10);
    c11 = wmma_bf16(a1, b1, c11);

    __syncthreads();                    // reads done; buffer may be overwritten next iter
    buf ^= 1;
  }

  gemm_epilogue_tile(c00, m0,      n0,      bias, outF, outBf, N, ldc, flags, lane);
  gemm_epilogue_tile(c01, m0,      n0 + 16, bias, outF, outBf, N, ldc, flags, lane);
  gemm_epilogue_tile(c10, m0 + 16, n0,      bias, outF, outBf, N, ldc, flags, lane);
  gemm_epilogue_tile(c11, m0 + 16, n0 + 16, bias, outF, outBf, N, ldc, flags, lane);
}

// ---------------- stem conv: 1 -> 32 ch, 3x3, stride 2, pad 1 ----------------
__global__ void stem_conv_kernel(const float* __restrict__ x, const float* __restrict__ w,
                                 const float* __restrict__ bias, float* __restrict__ out) {
  int idx = blockIdx.x * blockDim.x + threadIdx.x;           // B*32*64*1024 = 2^25
  int t = idx & (T1 - 1);
  int f = (idx >> 10) & 63;
  int c = (idx >> 16) & 31;
  int b = idx >> 21;
  float acc = bias[c];
#pragma unroll
  for (int df = 0; df < 3; ++df) {
    int hi = 2 * f + df - 1;
    if ((unsigned)hi >= F0) continue;
#pragma unroll
    for (int dt = 0; dt < 3; ++dt) {
      int wi = 2 * t + dt - 1;
      if ((unsigned)wi >= T0) continue;
      acc += x[((size_t)b * F0 + hi) * T0 + wi] * w[(c * 3 + df) * 3 + dt];
    }
  }
  out[idx] = acc;
}

// ---------------- CNN LayerNorm (over f axis, 64) + GELU ----------------
__global__ void cnn_ln_gelu_kernel(const float* __restrict__ x, const float* __restrict__ g,
                                   const float* __restrict__ be, float* __restrict__ out) {
  int idx = blockIdx.x * blockDim.x + threadIdx.x;           // B*32*1024 = 524288
  int t = idx & (T1 - 1);
  int c = (idx >> 10) & 31;
  int b = idx >> 15;
  size_t base = ((size_t)(b * 32 + c) * 64) * T1 + t;
  float s = 0.0f, s2 = 0.0f;
  for (int f = 0; f < 64; ++f) {
    float v = x[base + (size_t)f * T1];
    s += v; s2 += v * v;
  }
  float mean = s * (1.0f / 64.0f);
  float var  = s2 * (1.0f / 64.0f) - mean * mean;
  float rstd = rsqrtf(var + 1e-5f);
  for (int f = 0; f < 64; ++f) {
    size_t o = base + (size_t)f * T1;
    float y = (x[o] - mean) * rstd * g[f] + be[f];
    out[o] = gelu_f(y);
  }
}

// ---------------- residual conv: 32 -> 32 ch, 3x3, stride 1, pad 1 ----------------
__global__ __launch_bounds__(256) void res_conv_kernel(
    const float* __restrict__ x, const float* __restrict__ w, const float* __restrict__ bias,
    const float* __restrict__ residual, float* __restrict__ out) {
  __shared__ float ws[32 * 9];
  const int co = blockIdx.z & 31;
  const int b  = blockIdx.z >> 5;
  for (int i = threadIdx.x; i < 32 * 9; i += 256) ws[i] = w[co * 288 + i];
  __syncthreads();
  const int t = blockIdx.x * 256 + threadIdx.x;
  const int f = blockIdx.y;
  float acc = bias[co];
  for (int ci = 0; ci < 32; ++ci) {
    const float* xp = x + ((size_t)(b * 32 + ci) * 64) * T1;
    const float* wp = ws + ci * 9;
#pragma unroll
    for (int df = 0; df < 3; ++df) {
      int fi = f + df - 1;
      if ((unsigned)fi >= 64) continue;
      const float* row = xp + (size_t)fi * T1;
#pragma unroll
      for (int dt = 0; dt < 3; ++dt) {
        int ti = t + dt - 1;
        if ((unsigned)ti >= T1) continue;
        acc += row[ti] * wp[df * 3 + dt];
      }
    }
  }
  size_t o = ((size_t)(b * 32 + co) * 64 + f) * T1 + t;
  if (residual) acc += residual[o];
  out[o] = acc;
}

// ---------------- reshape (b,c,f,t) -> (b,t,c*f) with f32 -> bf16 ----------------
__global__ void tok_transpose_kernel(const float* __restrict__ x, __bf16* __restrict__ out) {
  int idx = blockIdx.x * blockDim.x + threadIdx.x;           // B*T1*CF = 2^25
  int cf = idx & (CF - 1);
  int t  = (idx >> 11) & (T1 - 1);
  int b  = idx >> 21;
  int c = cf >> 6, f = cf & 63;
  out[idx] = (__bf16)x[(((size_t)b * 32 + c) * 64 + f) * T1 + t];
}

// ---------------- token LayerNorm over 512 (one wave32 per token), bf16 out ----------------
__global__ __launch_bounds__(256) void token_ln_kernel(
    const float* __restrict__ x, const float* __restrict__ g, const float* __restrict__ be,
    __bf16* __restrict__ out) {
  const int wave = threadIdx.x >> 5;
  const int lane = threadIdx.x & 31;
  const int token = blockIdx.x * 8 + wave;
  const float* row = x + (size_t)token * DMODEL;
  float v[16], s = 0.0f, s2 = 0.0f;
#pragma unroll
  for (int i = 0; i < 16; ++i) {
    v[i] = row[lane + i * 32];
    s += v[i]; s2 += v[i] * v[i];
  }
#pragma unroll
  for (int off = 16; off > 0; off >>= 1) {
    s  += __shfl_xor(s, off);
    s2 += __shfl_xor(s2, off);
  }
  float mean = s * (1.0f / DMODEL);
  float var  = s2 * (1.0f / DMODEL) - mean * mean;
  float rstd = rsqrtf(var + 1e-5f);
#pragma unroll
  for (int i = 0; i < 16; ++i) {
    int d = lane + i * 32;
    float y = (v[i] - mean) * rstd * g[d] + be[d];
    out[(size_t)token * DMODEL + d] = (__bf16)y;
  }
}

// ---------------- minGRU Heinsen log-space scan (sequential over T per (b,d)) ----------------
__global__ void mingru_scan_kernel(const float* __restrict__ G, float* __restrict__ xs,
                                   __bf16* __restrict__ Hbf, int inner, int addDirect) {
  int idx = blockIdx.x * blockDim.x + threadIdx.x;           // BATCH * inner threads
  if (idx >= BATCH * inner) return;
  int d = idx % inner, b = idx / inner;
  const float* gp = G + (size_t)b * T1 * 2 * inner;
  float a = 0.0f, s = 0.0f;
  for (int t = 0; t < T1; ++t) {
    const float* rowp = gp + (size_t)t * 2 * inner;
    float h = rowp[d];
    float z = rowp[inner + d];
    float log_coeff = -softplus_f(z);               // log(1-z)
    float log_val   = -softplus_f(-z) + log_g_f(h); // log z + log g(h~)
    a += log_coeff;                                 // a_star (inclusive cumsum)
    float v = log_val - a;
    s = (t == 0) ? v : logaddexp_f(s, v);           // cumlogsumexp (inclusive)
    float outv = __expf(a + s);
    if (addDirect) xs[((size_t)b * T1 + t) * inner + d] += outv;   // layer 0: inner==512
    else           Hbf[((size_t)b * T1 + t) * inner + d] = (__bf16)outv;
  }
}

// ---------------- f32 -> bf16 elementwise ----------------
__global__ void f32_to_bf16_kernel(const float* __restrict__ in, __bf16* __restrict__ out, int n) {
  int i = blockIdx.x * blockDim.x + threadIdx.x;
  if (i < n) out[i] = (__bf16)in[i];
}

// ---------------- host-side launch helpers ----------------
static void launch_gemm(const __bf16* A, const __bf16* W, const float* bias,
                        float* outF, __bf16* outBf, int M, int N, int K, int ldc,
                        int flags, hipStream_t s) {
  dim3 grid((N + 127) / 128, M / 64);
  gemm_bf16_kernel<<<grid, 256, 0, s>>>(A, W, bias, outF, outBf, M, N, K, ldc, flags);
}
static void launch_cvt(const float* in, __bf16* out, int n, hipStream_t s) {
  f32_to_bf16_kernel<<<(n + 255) / 256, 256, 0, s>>>(in, out, n);
}

extern "C" void kernel_launch(void* const* d_in, const int* in_sizes, int n_in,
                              void* d_out, int out_size, void* d_ws, size_t ws_size,
                              hipStream_t stream) {
  (void)in_sizes; (void)n_in; (void)out_size; (void)ws_size;
  // ---- input map (setup_inputs flattening order) ----
  int ii = 0;
  const float* x      = (const float*)d_in[ii++];  // (16,1,128,2048)
  const float* cnn0_w = (const float*)d_in[ii++];
  const float* cnn0_b = (const float*)d_in[ii++];
  const float *ln1_g[3], *ln1_b[3], *c1_w[3], *c1_b[3], *ln2_g[3], *ln2_b[3], *c2_w[3], *c2_b[3];
  for (int i = 0; i < 3; ++i) {
    ln1_g[i] = (const float*)d_in[ii++]; ln1_b[i] = (const float*)d_in[ii++];
    c1_w[i]  = (const float*)d_in[ii++]; c1_b[i]  = (const float*)d_in[ii++];
    ln2_g[i] = (const float*)d_in[ii++]; ln2_b[i] = (const float*)d_in[ii++];
    c2_w[i]  = (const float*)d_in[ii++]; c2_b[i]  = (const float*)d_in[ii++];
  }
  const float *norm_g[5], *norm_b[5], *hg_w[5], *out_w[5];
  for (int i = 0; i < 5; ++i) {
    norm_g[i] = (const float*)d_in[ii++];
    norm_b[i] = (const float*)d_in[ii++];
    hg_w[i]   = (const float*)d_in[ii++];
    out_w[i]  = (i == 0) ? nullptr : (const float*)d_in[ii++];  // layer0 out_w is None
  }
  const float* fc_w   = (const float*)d_in[ii++];
  const float* fc_b   = (const float*)d_in[ii++];
  const float* clf1_w = (const float*)d_in[ii++];
  const float* clf1_b = (const float*)d_in[ii++];
  const float* clf2_w = (const float*)d_in[ii++];
  const float* clf2_b = (const float*)d_in[ii++];
  float* out = (float*)d_out;                       // (16,1024,29) f32

  // ---- workspace layout ----
  const size_t MB = 1ull << 20;
  char* ws = (char*)d_ws;
  float*  bufA  = (float*)(ws);                     // 128 MiB CNN ping  (later: x_stream etc.)
  float*  bufB  = (float*)(ws + 128 * MB);          // 128 MiB CNN pong  (later: tokens/H)
  float*  bufT  = (float*)(ws + 256 * MB);          // 128 MiB LN temp   (later: G)
  __bf16* wbf   = (__bf16*)(ws + 384 * MB);         // 8 MiB weight bf16 scratch
  // aliases (CNN buffers dead by the time these are live)
  __bf16* tokbf = (__bf16*)bufB;                    // 64 MiB  (16384 x 2048 bf16)
  float*  xs    = (float*)bufA;                     // 32 MiB  (16384 x 512 f32)
  __bf16* xbf   = (__bf16*)(ws + 32 * MB);          // 16 MiB
  __bf16* nbf   = (__bf16*)(ws + 48 * MB);          // 16 MiB
  float*  G     = bufT;                             // up to 128 MiB (16384 x 2048 f32)
  __bf16* Hbf   = (__bf16*)(ws + 192 * MB);         // 32 MiB  (16384 x 1024 bf16)

  const int NCNN = BATCH * C1 * F1 * T1;            // 2^25

  // ---- stem conv ----
  stem_conv_kernel<<<NCNN / 256, 256, 0, stream>>>(x, cnn0_w, cnn0_b, bufA);

  // ---- 3 residual CNN blocks (activation lives in bufA) ----
  for (int i = 0; i < 3; ++i) {
    cnn_ln_gelu_kernel<<<(BATCH * C1 * T1) / 256, 256, 0, stream>>>(bufA, ln1_g[i], ln1_b[i], bufT);
    res_conv_kernel<<<dim3(T1 / 256, F1, BATCH * C1), 256, 0, stream>>>(bufT, c1_w[i], c1_b[i], nullptr, bufB);
    cnn_ln_gelu_kernel<<<(BATCH * C1 * T1) / 256, 256, 0, stream>>>(bufB, ln2_g[i], ln2_b[i], bufT);
    res_conv_kernel<<<dim3(T1 / 256, F1, BATCH * C1), 256, 0, stream>>>(bufT, c2_w[i], c2_b[i], bufA, bufA);
  }

  // ---- reshape to tokens + bf16 ----
  tok_transpose_kernel<<<NCNN / 256, 256, 0, stream>>>(bufA, tokbf);

  // ---- fc: (16384 x 2048) x (2048 -> 512) + bias ----
  launch_cvt(fc_w, wbf, DMODEL * CF, stream);
  launch_gemm(tokbf, wbf, fc_b, xs, nullptr, M_TOK, DMODEL, CF, DMODEL, 0, stream);

  // ---- 5 minGRU layers ----
  for (int i = 0; i < 5; ++i) {
    const int inner = (i == 0) ? DMODEL : 2 * DMODEL;
    token_ln_kernel<<<M_TOK / 8, 256, 0, stream>>>(xs, norm_g[i], norm_b[i], nbf);
    launch_cvt(hg_w[i], wbf, 2 * inner * DMODEL, stream);
    launch_gemm(nbf, wbf, nullptr, G, nullptr, M_TOK, 2 * inner, DMODEL, 2 * inner, 0, stream);
    const int nscan = BATCH * inner;
    mingru_scan_kernel<<<(nscan + 255) / 256, 256, 0, stream>>>(
        G, xs, Hbf, inner, (i == 0) ? 1 : 0);
    if (i > 0) {
      launch_cvt(out_w[i], wbf, DMODEL * inner, stream);
      launch_gemm(Hbf, wbf, nullptr, xs, nullptr, M_TOK, DMODEL, inner, DMODEL, /*residual*/2, stream);
    }
  }

  // ---- classifier ----
  launch_cvt(xs, xbf, M_TOK * DMODEL, stream);
  launch_cvt(clf1_w, wbf, DMODEL * DMODEL, stream);
  launch_gemm(xbf, wbf, clf1_b, nullptr, nbf, M_TOK, DMODEL, DMODEL, DMODEL, /*gelu*/1, stream);
  launch_cvt(clf2_w, wbf, NCLASS * DMODEL, stream);
  launch_gemm(nbf, wbf, clf2_b, out, nullptr, M_TOK, NCLASS, DMODEL, NCLASS, 0, stream);
}
